// ASR_50775103373962
// MI455X (gfx1250) — matrix operand order repaired and checked by reading, not compiled, not run on hardware
//
#include <hip/hip_runtime.h>

typedef _Float16 h16_t;
typedef __attribute__((ext_vector_type(16))) _Float16 v16h;
typedef __attribute__((ext_vector_type(8)))  float    v8f;
typedef __attribute__((ext_vector_type(2)))  _Float16 h2v;

#define TSTEPS 600
#define BATCH  32
#define FIN    120
#define FINP   128
#define ENCH   512
#define ENCOUT 1024
#define DECH   512
#define VOCAB  4096
#define DSTEPS 120
#define KENC   640     // 128 (padded input) + 512 (hidden)
#define KDEC   2048    // 512 last + 1024 ctx + 512 hidden

__device__ __forceinline__ float sigf(float x){ return 1.0f/(1.0f + __expf(-x)); }

// ---- WMMA fragment loaders (wave32; contiguous k order, same for A and B) ----
// A 16x32: lane row m=lane%16; half=lane/16; v[0..15]: K = half*16 + {0..15}
__device__ __forceinline__ v16h frag_a(const h16_t* p, int lh){
  v16h a;
  const h16_t* q0 = p + lh*16;
  const h16_t* q1 = p + lh*16 + 8;
#pragma unroll
  for (int j=0;j<4;++j){
    h2v t0 = *(const h2v*)(q0 + 2*j);
    h2v t1 = *(const h2v*)(q1 + 2*j);
    a[2*j]   = t0[0]; a[2*j+1] = t0[1];
    a[8+2*j] = t1[0]; a[9+2*j] = t1[1];
  }
  return a;
}
// B 32x16: lane col n=lane%16; half=lane/16; v[0..15]: K = half*16 + {0..15}
// We store B as W[N][K] row-major (out = act @ W^T), so column n == W row n.
__device__ __forceinline__ v16h frag_b(const h16_t* p, int lh){
  v16h b;
  const h16_t* q = p + lh*16;
#pragma unroll
  for (int j=0;j<8;++j){
    h2v t = *(const h2v*)(q + 2*j);
    b[2*j] = t[0]; b[2*j+1] = t[1];
  }
  return b;
}
__device__ __forceinline__ v8f wmma32(v16h a, v16h b, v8f c){
  return __builtin_amdgcn_wmma_f32_16x16x32_f16(false, a, false, b, (short)0, c, false, false);
}
__device__ __forceinline__ v8f vzero8(){
  v8f z = {0.f,0.f,0.f,0.f,0.f,0.f,0.f,0.f};
  return z;
}

// ================= weight packing =================
__global__ __launch_bounds__(256) void cvt_f16_kernel(const float* __restrict__ s,
                                                      h16_t* __restrict__ d, int n){
  int i = blockIdx.x*256 + threadIdx.x;
  if (i < n) d[i] = (h16_t)s[i];
}

__global__ __launch_bounds__(256) void pack_enc_kernel(
    const float* __restrict__ Wih_f, const float* __restrict__ Whh_f,
    const float* __restrict__ Wih_b, const float* __restrict__ Whh_b,
    h16_t* __restrict__ Wcat){            // [2][2048][KENC]
  int i = blockIdx.x*256 + threadIdx.x;
  const int per = 2048*KENC;
  if (i >= 2*per) return;
  int dir = i/per, r = i%per, n = r/KENC, k = r%KENC;
  const float* Wih = dir ? Wih_b : Wih_f;
  const float* Whh = dir ? Whh_b : Whh_f;
  float v = (k < FIN) ? Wih[n*FIN + k] : (k < FINP ? 0.f : Whh[n*ENCH + (k-FINP)]);
  Wcat[i] = (h16_t)v;
}

__global__ __launch_bounds__(256) void pack_dec_kernel(
    const float* __restrict__ Wih_d, const float* __restrict__ Whh_d,
    h16_t* __restrict__ Wcat){            // [2048][KDEC]
  int i = blockIdx.x*256 + threadIdx.x;
  if (i >= 2048*KDEC) return;
  int n = i/KDEC, k = i%KDEC;
  float v = (k < 1536) ? Wih_d[n*1536 + k] : Whh_d[n*DECH + (k-1536)];
  Wcat[i] = (h16_t)v;
}

__global__ __launch_bounds__(256) void dec_init_kernel(
    float* __restrict__ cbuf, h16_t* __restrict__ hdec,
    h16_t* __restrict__ xin, const h16_t* __restrict__ E16){
  int i = blockIdx.x*256 + threadIdx.x;
  if (i < BATCH*DECH){ cbuf[i] = 0.f; hdec[i] = (h16_t)0.f; }
  if (i < BATCH*KDEC){
    int k = i & (KDEC-1);
    xin[i] = (k < DECH) ? E16[k] : (h16_t)0.f;  // last0 = E[0], ctx/h = 0
  }
}

// ================= persistent bidirectional LSTM encoder =================
// grid = 2 blocks (dir), block = 1024 threads = 32 waves.
// Each wave owns 2 jobs; job = (mtile in 0..1, ctile in 0..31); per job it
// computes the 4 gate tiles (i,f,g,o) for 16 hidden cols x 16 batch rows and
// keeps c in registers across all 600 steps.
__global__ __launch_bounds__(1024) void encoder_kernel(
    const float* __restrict__ x,          // [B,T,FIN]
    const h16_t* __restrict__ Wcat,       // [2][2048][KENC]
    const float* __restrict__ b_f, const float* __restrict__ b_b,
    h16_t* __restrict__ enc){             // [B,T,ENCOUT] f16
  const int dir = blockIdx.x;
  const float* bias = dir ? b_b : b_f;
  const h16_t* W = Wcat + (size_t)dir*2048*KENC;

  __shared__ h16_t hbuf[BATCH][ENCH];
  __shared__ h16_t xbuf[BATCH][FINP];

  const int tid = threadIdx.x;
  for (int i=tid; i<BATCH*ENCH; i+=1024) ((h16_t*)hbuf)[i] = (h16_t)0.f;
  for (int i=tid; i<BATCH*FINP; i+=1024) ((h16_t*)xbuf)[i] = (h16_t)0.f;

  const int wid = tid>>5, lane = tid&31, lm = lane&15, lh = lane>>4;
  const int job0 = wid*2;

  v8f cst[2]; cst[0] = vzero8(); cst[1] = vzero8();

  for (int t=0; t<TSTEPS; ++t){
    const int tt = dir ? (TSTEPS-1-t) : t;
    __syncthreads();                       // prev-step hbuf writes visible / xbuf free
    for (int i=tid; i<BATCH*FIN; i+=1024){
      int b = i/FIN, k = i%FIN;
      xbuf[b][k] = (h16_t)x[((size_t)b*TSTEPS + tt)*FIN + k];
    }
    __syncthreads();                       // xbuf + hbuf ready

    v8f acc[2][4];
#pragma unroll
    for (int j=0;j<2;++j)
#pragma unroll
      for (int g=0;g<4;++g) acc[j][g] = vzero8();

#pragma unroll
    for (int j=0;j<2;++j){
      const int job = job0 + j, mt = job&1, ct = job>>1;
      const int arow = mt*16 + lm;
      const int wrow = ct*16 + lm;
      for (int kc=0; kc<KENC/32; ++kc){
        const int kb = kc*32;
        const h16_t* pA = (kb < FINP) ? &xbuf[arow][kb] : &hbuf[arow][kb-FINP];
        v16h a = frag_a(pA, lh);
#pragma unroll
        for (int g=0; g<4; ++g){
          const h16_t* pB = W + (size_t)(g*ENCH + wrow)*KENC + kb;
          v16h bfr = frag_b(pB, lh);
          acc[j][g] = wmma32(a, bfr, acc[j][g]);
        }
      }
    }
    __syncthreads();                       // all hbuf reads done

#pragma unroll
    for (int j=0;j<2;++j){
      const int job = job0 + j, mt = job&1, ct = job>>1;
      const int col = ct*16 + lm;
      const float bi = bias[col], bf = bias[ENCH+col],
                  bg = bias[2*ENCH+col], bo = bias[3*ENCH+col];
#pragma unroll
      for (int r=0;r<8;++r){
        const int row = mt*16 + r + lh*8;  // batch index
        float iv = acc[j][0][r] + bi;
        float fv = acc[j][1][r] + bf;
        float gv = acc[j][2][r] + bg;
        float ov = acc[j][3][r] + bo;
        float c  = sigf(fv)*cst[j][r] + sigf(iv)*tanhf(gv);
        cst[j][r] = c;
        h16_t hh = (h16_t)(sigf(ov)*tanhf(c));
        hbuf[row][col] = hh;
        enc[((size_t)row*TSTEPS + tt)*ENCOUT + dir*ENCH + col] = hh;
      }
    }
  }
}

// ================= generic WMMA GEMM: out = act( A[M,K] @ W[N,K]^T + bias ) ==
// one wave per 16x16 output tile
template<int ACT, int OUT16>
__global__ __launch_bounds__(256) void gemm_kernel(
    const h16_t* __restrict__ A, const h16_t* __restrict__ W,
    const float* __restrict__ bias, void* __restrict__ out,
    int N, int K, size_t ostride){
  const int gw   = (blockIdx.x*256 + threadIdx.x)>>5;
  const int lane = threadIdx.x & 31, lm = lane&15, lh = lane>>4;
  const int ntiles = N >> 4;
  const int mt = gw / ntiles, nt = gw % ntiles;
  const h16_t* pa = A + (size_t)(mt*16 + lm)*K;
  const h16_t* pw = W + (size_t)(nt*16 + lm)*K;
  v8f acc = vzero8();
  for (int kb=0; kb<K; kb+=32){
    v16h a = frag_a(pa + kb, lh);
    v16h b = frag_b(pw + kb, lh);
    acc = wmma32(a, b, acc);
  }
  const int col = nt*16 + lm;
  const float bv = bias[col];
#pragma unroll
  for (int r=0;r<8;++r){
    const int row = mt*16 + r + lh*8;
    float v = acc[r] + bv;
    if (ACT) v = tanhf(v);
    if (OUT16) ((h16_t*)out)[(size_t)row*ostride + col] = (h16_t)v;
    else       ((float*)out)[(size_t)row*ostride + col] = v;
  }
}

// ================= decoder: scores + softmax + context (one block per batch) ==
__global__ __launch_bounds__(1024) void scores_ctx_kernel(
    const h16_t* __restrict__ q16,   // [B,DECH]
    const h16_t* __restrict__ key16, // [B*T,512]
    const h16_t* __restrict__ val16, // [B*T,1024]
    const int* __restrict__ flen,
    float* __restrict__ att_seq,     // + step*600 ; element (b,t) at b*120*600 + t
    const h16_t* __restrict__ hdec,  // h_prev -> copy into xin[1536:2048]
    h16_t* __restrict__ xin){        // [B,KDEC]
  const int b = blockIdx.x, tid = threadIdx.x;
  __shared__ float qs[DECH];
  __shared__ float attn[TSTEPS];
  __shared__ float red[1024];

  if (tid < DECH){
    qs[tid] = (float)q16[(size_t)b*DECH + tid];
    xin[(size_t)b*KDEC + 1536 + tid] = hdec[(size_t)b*DECH + tid]; // stage h_prev
  }
  const int len = flen[b];
  __syncthreads();

  float s = -1.0e38f;
  if (tid < TSTEPS){
    const h16_t* kr = key16 + ((size_t)b*TSTEPS + tid)*512;
    float acc = 0.f;
#pragma unroll 4
    for (int d=0; d<512; d+=2){
      h2v kp = *(const h2v*)(kr + d);
      acc += qs[d]*(float)kp[0] + qs[d+1]*(float)kp[1];
    }
    s = acc * 2.0f;                         // / TEMP (0.5)
    if (tid >= len) s = -1.0e30f;           // mask
  }
  red[tid] = s; __syncthreads();
  for (int o=512;o>0;o>>=1){ if (tid<o) red[tid]=fmaxf(red[tid],red[tid+o]); __syncthreads(); }
  const float mx = red[0]; __syncthreads();

  float e = (tid < TSTEPS) ? __expf(s - mx) : 0.f;
  red[tid] = e; __syncthreads();
  for (int o=512;o>0;o>>=1){ if (tid<o) red[tid]+=red[tid+o]; __syncthreads(); }
  const float inv = 1.0f/red[0];
  if (tid < TSTEPS){
    float p = e*inv;
    attn[tid] = p;
    att_seq[(size_t)b*DSTEPS*TSTEPS + tid] = p;
  }
  __syncthreads();

  // ctx: thread = output channel (coalesced val reads)
  const int w = tid;                         // 0..1023
  float acc = 0.f;
  const h16_t* vb = val16 + (size_t)b*TSTEPS*ENCOUT + w;
  for (int t=0; t<TSTEPS; ++t) acc += attn[t]*(float)vb[(size_t)t*ENCOUT];
  xin[(size_t)b*KDEC + DECH + w] = (h16_t)acc;
}

// ================= decoder LSTM cell: gates + c/h update ====================
// 64 waves; wave = (mtile, hcol tile); computes 4 gate tiles, updates c,h.
__global__ __launch_bounds__(256) void dec_cell_kernel(
    const h16_t* __restrict__ xin,  // [B,KDEC] = [last | ctx | h_prev]
    const h16_t* __restrict__ Wd,   // [2048][KDEC]
    const float* __restrict__ bd,
    float* __restrict__ cbuf,       // [B,DECH]
    h16_t* __restrict__ hdec){      // [B,DECH]
  const int gw = (blockIdx.x*256 + threadIdx.x)>>5;
  const int lane = threadIdx.x & 31, lm = lane&15, lh = lane>>4;
  const int mt = gw & 1, ct = gw >> 1;
  const h16_t* pa = xin + (size_t)(mt*16 + lm)*KDEC;
  const int wrow = ct*16 + lm;
  v8f acc[4];
#pragma unroll
  for (int g=0;g<4;++g) acc[g] = vzero8();
  for (int kb=0; kb<KDEC; kb+=32){
    v16h a = frag_a(pa + kb, lh);
#pragma unroll
    for (int g=0;g<4;++g){
      v16h b = frag_b(Wd + (size_t)(g*DECH + wrow)*KDEC + kb, lh);
      acc[g] = wmma32(a, b, acc[g]);
    }
  }
  const int col = ct*16 + lm;
  const float bi = bd[col], bf = bd[DECH+col], bg = bd[2*DECH+col], bo = bd[3*DECH+col];
#pragma unroll
  for (int r=0;r<8;++r){
    const int row = mt*16 + r + lh*8;
    float iv = acc[0][r]+bi, fv = acc[1][r]+bf, gv = acc[2][r]+bg, ov = acc[3][r]+bo;
    float c = sigf(fv)*cbuf[row*DECH + col] + sigf(iv)*tanhf(gv);
    cbuf[row*DECH + col] = c;
    hdec[row*DECH + col] = (h16_t)(sigf(ov)*tanhf(c));
  }
}

// ================= greedy argmax + embedding gather =========================
__global__ __launch_bounds__(256) void argmax_embed_kernel(
    const float* __restrict__ logits,  // base + step*VOCAB; row stride DSTEPS*VOCAB
    const h16_t* __restrict__ E16, h16_t* __restrict__ xin){
  const int b = blockIdx.x, tid = threadIdx.x;
  const float* row = logits + (size_t)b*DSTEPS*VOCAB;
  float bv = -1.0e38f; int bi = VOCAB;
  for (int v=tid; v<VOCAB; v+=256){
    float xv = row[v];
    if (xv > bv || (xv == bv && v < bi)){ bv = xv; bi = v; }
  }
  __shared__ float sv[256]; __shared__ int si[256];
  sv[tid] = bv; si[tid] = bi; __syncthreads();
  for (int o=128;o>0;o>>=1){
    if (tid < o){
      float ov = sv[tid+o]; int oi = si[tid+o];
      if (ov > sv[tid] || (ov == sv[tid] && oi < si[tid])){ sv[tid]=ov; si[tid]=oi; }
    }
    __syncthreads();
  }
  const int best = si[0];
  for (int k=tid; k<DECH; k+=256)
    xin[(size_t)b*KDEC + k] = E16[(size_t)best*DECH + k];
}

// ======================= host orchestration =================================
extern "C" void kernel_launch(void* const* d_in, const int* in_sizes, int n_in,
                              void* d_out, int out_size, void* d_ws, size_t ws_size,
                              hipStream_t stream){
  (void)in_sizes; (void)n_in; (void)out_size; (void)ws_size;
  const float* x     = (const float*)d_in[0];
  const float* Wih_f = (const float*)d_in[1];
  const float* Whh_f = (const float*)d_in[2];
  const float* b_f   = (const float*)d_in[3];
  const float* Wih_b = (const float*)d_in[4];
  const float* Whh_b = (const float*)d_in[5];
  const float* b_b   = (const float*)d_in[6];
  const float* Wq    = (const float*)d_in[7];
  const float* bq    = (const float*)d_in[8];
  const float* Wk    = (const float*)d_in[9];
  const float* bk    = (const float*)d_in[10];
  const float* Wv    = (const float*)d_in[11];
  const float* bv    = (const float*)d_in[12];
  const float* Wih_d = (const float*)d_in[13];
  const float* Whh_d = (const float*)d_in[14];
  const float* b_d   = (const float*)d_in[15];
  const float* Wc    = (const float*)d_in[16];
  const float* bc    = (const float*)d_in[17];
  const float* E     = (const float*)d_in[18];
  const int*   flen  = (const int*)d_in[19];

  char* ws = (char*)d_ws; size_t off = 0;
  auto take = [&](size_t bytes)->char*{
    char* p = ws + off; off = (off + bytes + 255) & ~(size_t)255; return p; };

  h16_t* WcatE = (h16_t*)take((size_t)2*2048*KENC*2);
  h16_t* Wq16  = (h16_t*)take((size_t)512*512*2);
  h16_t* Wk16  = (h16_t*)take((size_t)512*1024*2);
  h16_t* Wv16  = (h16_t*)take((size_t)1024*1024*2);
  h16_t* WcatD = (h16_t*)take((size_t)2048*KDEC*2);
  h16_t* Wc16  = (h16_t*)take((size_t)VOCAB*512*2);
  h16_t* E16   = (h16_t*)take((size_t)VOCAB*512*2);
  h16_t* enc16 = (h16_t*)take((size_t)BATCH*TSTEPS*ENCOUT*2);
  h16_t* key16 = (h16_t*)take((size_t)BATCH*TSTEPS*512*2);
  h16_t* val16 = (h16_t*)take((size_t)BATCH*TSTEPS*1024*2);
  h16_t* hdec  = (h16_t*)take((size_t)BATCH*DECH*2);
  float* cbuf  = (float*)take((size_t)BATCH*DECH*4);
  h16_t* q16   = (h16_t*)take((size_t)BATCH*DECH*2);
  h16_t* xin   = (h16_t*)take((size_t)BATCH*KDEC*2);

  float* att_out = (float*)d_out;                                   // [B,120,4096]
  float* att_seq = (float*)d_out + (size_t)BATCH*DSTEPS*VOCAB;      // [B,120,600]

  // --- pack weights to f16 ---
  pack_enc_kernel<<<(2*2048*KENC+255)/256, 256, 0, stream>>>(Wih_f,Whh_f,Wih_b,Whh_b,WcatE);
  cvt_f16_kernel<<<(512*512+255)/256,   256, 0, stream>>>(Wq, Wq16, 512*512);
  cvt_f16_kernel<<<(512*1024+255)/256,  256, 0, stream>>>(Wk, Wk16, 512*1024);
  cvt_f16_kernel<<<(1024*1024+255)/256, 256, 0, stream>>>(Wv, Wv16, 1024*1024);
  cvt_f16_kernel<<<(VOCAB*512+255)/256, 256, 0, stream>>>(Wc, Wc16, VOCAB*512);
  cvt_f16_kernel<<<(VOCAB*512+255)/256, 256, 0, stream>>>(E,  E16,  VOCAB*512);
  pack_dec_kernel<<<(2048*KDEC+255)/256, 256, 0, stream>>>(Wih_d, Whh_d, WcatD);

  // --- encoder: both directions, whole sequence, one launch ---
  encoder_kernel<<<2, 1024, 0, stream>>>(x, WcatE, b_f, b_b, enc16);

  // --- attention key / value projections (tanh) ---
  { const int tiles = (BATCH*TSTEPS/16)*(512/16);   // 38400
    gemm_kernel<1,1><<<tiles/8, 256, 0, stream>>>(enc16, Wk16, bk, key16, 512, ENCOUT, 512); }
  { const int tiles = (BATCH*TSTEPS/16)*(1024/16);  // 76800
    gemm_kernel<1,1><<<tiles/8, 256, 0, stream>>>(enc16, Wv16, bv, val16, 1024, ENCOUT, 1024); }

  // --- decoder init: c=h=0, xin = [E[0] | 0 | 0] ---
  dec_init_kernel<<<(BATCH*KDEC+255)/256, 256, 0, stream>>>(cbuf, hdec, xin, E16);

  for (int s=0; s<DSTEPS; ++s){
    // q = tanh(h @ Wq^T + bq)
    gemm_kernel<1,1><<<8, 256, 0, stream>>>(hdec, Wq16, bq, q16, 512, 512, 512);
    // scores -> softmax -> ctx ; also stages h_prev into xin
    scores_ctx_kernel<<<BATCH, 1024, 0, stream>>>(
        q16, key16, val16, flen, att_seq + (size_t)s*TSTEPS, hdec, xin);
    // gates + LSTM cell
    dec_cell_kernel<<<8, 256, 0, stream>>>(xin, WcatD, b_d, cbuf, hdec);
    // logits straight into d_out (f32), row stride 120*4096
    gemm_kernel<0,0><<<64, 256, 0, stream>>>(
        hdec, Wc16, bc, (void*)(att_out + (size_t)s*VOCAB), VOCAB, 512,
        (size_t)DSTEPS*VOCAB);
    // greedy argmax + embedding for next step's xin[:,0:512]
    argmax_embed_kernel<<<BATCH, 256, 0, stream>>>(
        att_out + (size_t)s*VOCAB, E16, xin);
  }
}
